// NRMS_17695265260020
// MI455X (gfx1250) — matrix operand order repaired and checked by reading, not compiled, not run on hardware
//
#include <hip/hip_runtime.h>
#include <math.h>
#include <stdint.h>

// ---------------------------------------------------------------------------
// NRMS forward for MI455X (gfx1250, wave32, WMMA + async LDS copies).
// Pipeline:
//   P0: pack weights f32 -> bf16 slabs in ws (once; enables async staging)
//   K1: encode<30,32,GATHER> over B*NH=6400 blocks  -> news_enc [B,50,300]
//   K2: encode<30,32,GATHER> over B*NC=640  blocks  -> cand_enc [B,5,300]
//   K3: encode<50,64,!GATHER> over B=128    blocks  -> user_rep [B,300]
//   K4: score kernel: softmax(user_rep . cand_enc)  -> d_out   [B,5]
// Big GEMMs: v_wmma_f32_16x16x32_bf16. Weight slabs staged into LDS with
// global_load_async_to_lds_b128 (+ s_wait_asynccnt) when ws is big enough,
// else an f32->bf16 conversion fallback. Attention math is VALU (3% of FLOPs).
// ---------------------------------------------------------------------------

typedef __attribute__((ext_vector_type(16))) __bf16        v16bf;
typedef __attribute__((ext_vector_type(8)))  float         v8f;
typedef __attribute__((ext_vector_type(4)))  unsigned int  u32x4;

union V16 {
    u32x4 u[2];
    v16bf v;
};

#define NTHREADS 256
#define NWAVES   8
#define DPAD     320   // K dim (300) padded to 10 steps of 32
#define HEADS    15
#define DKV      20    // per-head dim
#define AHID     200   // additive attention hidden
#define HD       300   // model dim

__device__ __forceinline__ unsigned short f2bfu(float f) {
    unsigned u = __builtin_bit_cast(unsigned, f);
    unsigned r = u + 0x7FFFu + ((u >> 16) & 1u);   // round-to-nearest-even
    return (unsigned short)(r >> 16);
}
__device__ __forceinline__ float bf2f(unsigned short h) {
    unsigned u = ((unsigned)h) << 16;
    return __builtin_bit_cast(float, u);
}

// ---------------------------------------------------------------------------
// Weight pre-pack: W[300][N] f32 -> [10 ksteps][NP cols][40 rows] bf16,
// i.e. exactly the LDS slab layout the GEMM consumes, so staging becomes a
// straight async memcpy. NP = ceil(N/16)*16; rows beyond 300 / cols beyond N
// are zero.
// ---------------------------------------------------------------------------
template <int N>
__global__ __launch_bounds__(NTHREADS) void pack_kernel(
    const float* __restrict__ W, unsigned short* __restrict__ out, int nmat)
{
    constexpr int NT   = (N + 15) / 16;
    constexpr int NP   = NT * 16;
    constexpr int SLAB = NP * 40;          // ushorts per k-slab
    const int blk = blockIdx.x;
    const int mat = blk / 10, kk = blk - mat * 10;
    const float* Wm = W + (size_t)mat * 300 * N;
    unsigned short* o = out + (size_t)mat * 10 * SLAB + (size_t)kk * SLAB;
    for (int idx = threadIdx.x; idx < 32 * NP; idx += NTHREADS) {
        int r  = idx / NP;
        int c  = idx - r * NP;             // consecutive tid -> consecutive c
        int gr = kk * 32 + r;
        float w = (gr < 300 && c < N) ? Wm[(size_t)gr * N + c] : 0.f;
        o[c * 40 + r] = f2bfu(w);
    }
}

// ---------------------------------------------------------------------------
// Tiled GEMM: D[Lvalid x N] = X[LP x 320](bf16, LDS) @ W[300 x N].
// Per k-step the 32xNP weight slab lands in LDS either via
// global_load_async_to_lds_b128 from the pre-packed bf16 image (fast path),
// or via f32 load + convert (fallback). A/B fragments follow the CDNA5
// 16-bit A 16x32 / B 32x16 VGPR striping; B fragments are contiguous because
// the slab is stored K-major per column.
// ---------------------------------------------------------------------------
template <int LP, int N>
__device__ __forceinline__ void gemm_bf16(
    const unsigned short* __restrict__ sX,   // [LP][320] bf16 (zero padded)
    unsigned short* __restrict__ sWt,        // [NP][40] bf16 staging (16B aligned)
    const float* __restrict__ Wg,            // [300][N] row-major f32
    const unsigned short* __restrict__ Wpk,  // packed bf16 slabs or nullptr
    unsigned short* __restrict__ outB,       // bf16 out (or nullptr)
    float* __restrict__ outF,                // f32  out (or nullptr)
    int outStride, int Lvalid, int tid)
{
    constexpr int NT = (N + 15) / 16;        // n tiles
    constexpr int NP = NT * 16;
    constexpr int SLABB = NP * 40 * 2;       // slab bytes
    constexpr int NCHUNK = SLABB / 16;       // b128 chunks per slab
    constexpr int MT = LP / 16;              // m tiles
    constexpr int TT = MT * NT;              // total tiles
    constexpr int PW = (TT + NWAVES - 1) / NWAVES;

    const int wave   = __builtin_amdgcn_readfirstlane(tid >> 5);  // scalar
    const int lane   = tid & 31;
    const int lane16 = lane & 15;
    const int hi8    = (lane >= 16) ? 8 : 0;   // A k-shift / C row shift
    const int hi16   = (lane >= 16) ? 16 : 0;  // unused (B k-shift is hi8 too)

    const unsigned sWtBase = (unsigned)(uintptr_t)(const void*)sWt;
    const unsigned long long pkBase = (unsigned long long)(uintptr_t)Wpk;

    v8f acc[PW];
#pragma unroll
    for (int i = 0; i < PW; ++i)
#pragma unroll
        for (int r = 0; r < 8; ++r) acc[i][r] = 0.f;

    for (int kk = 0; kk < 10; ++kk) {
        __syncthreads();
        if (Wpk) {
            // Fast path: async bf16 slab copy, 16B per lane per issue.
            for (int idx = tid; idx < NCHUNK; idx += NTHREADS) {
                unsigned lds  = sWtBase + (unsigned)(idx * 16);
                unsigned goff = (unsigned)(kk * SLABB + idx * 16);
                asm volatile("global_load_async_to_lds_b128 %0, %1, %2"
                             :: "v"(lds), "v"(goff), "s"(pkBase)
                             : "memory");
            }
            if (kk + 1 < 10) {  // pull next slab toward L2/L0
                __builtin_prefetch((const char*)Wpk + (kk + 1) * SLABB + tid * 64, 0, 0);
            }
            asm volatile("s_wait_asynccnt 0" ::: "memory");
        } else {
            // Fallback: stage + transpose + convert from f32 weights.
            for (int idx = tid; idx < 32 * NP; idx += NTHREADS) {
                int r  = idx / NP;
                int c  = idx - r * NP;
                int gr = kk * 32 + r;
                float w = (gr < 300 && c < N) ? Wg[(size_t)gr * N + c] : 0.f;
                sWt[c * 40 + r] = f2bfu(w);
            }
        }
        __syncthreads();
#pragma unroll
        for (int i = 0; i < PW; ++i) {
            int t = wave + i * NWAVES;         // scalar -> s_cbranch, EXEC untouched
            if (t < TT) {
                int mt = t / NT, nt = t - mt * NT;
                // A: row = mt*16+lane16; lanes<16: K {0..7,16..23}, hi: {8..15,24..31}
                const unsigned short* ap = sX + (mt * 16 + lane16) * DPAD + kk * 32 + hi8;
                V16 a;
                a.u[0] = *(const u32x4*)ap;
                a.u[1] = *(const u32x4*)(ap + 16);
                // B: col = nt*16+lane16; lanes<16: K {0..7,16..23}, hi: {8..15,24..31}
                const unsigned short* bp_ = sWt + (nt * 16 + lane16) * 40 + hi8;
                V16 b;
                b.u[0] = *(const u32x4*)bp_;
                b.u[1] = *(const u32x4*)(bp_ + 16);
                acc[i] = __builtin_amdgcn_wmma_f32_16x16x32_bf16(
                    false, a.v, false, b.v, (short)0, acc[i], false, false);
            }
        }
    }
    __syncthreads();
    // D: VGPR r, lanes<16 -> (M=r, N=lane16), lanes>=16 -> (M=r+8, N=lane16)
#pragma unroll
    for (int i = 0; i < PW; ++i) {
        int t = wave + i * NWAVES;
        if (t < TT) {
            int mt  = t / NT, nt = t - mt * NT;
            int col = nt * 16 + lane16;
            if (col < N) {
#pragma unroll
                for (int r = 0; r < 8; ++r) {
                    int row = mt * 16 + r + hi8;
                    if (row < Lvalid) {
                        if (outB) outB[row * outStride + col] = f2bfu(acc[i][r]);
                        else      outF[row * outStride + col] = acc[i][r];
                    }
                }
            }
        }
    }
    __syncthreads();
}

// ---------------------------------------------------------------------------
// One block = one (b, p) NRMS encoder.
// ---------------------------------------------------------------------------
template <int LVAL, int LP, bool GATHER>
__global__ __launch_bounds__(NTHREADS) void encode_kernel(
    const int*   __restrict__ tokens,  // [nblk, LVAL] (GATHER only)
    const float* __restrict__ xin,     // [B, LVAL, 300] (!GATHER only)
    const float* __restrict__ emb,     // [V, 300]
    const float* __restrict__ Wq, const float* __restrict__ Wk,
    const float* __restrict__ Wv, const float* __restrict__ Wp,
    const float* __restrict__ bp, const float* __restrict__ qv,
    const unsigned short* pWq, const unsigned short* pWk,
    const unsigned short* pWv, const unsigned short* pWp,   // packed (or null)
    int P, int sQKV, int sP, int sB, int sPkQKV, int sPkP,
    float* __restrict__ out)           // [nblk, 300]
{
    const int blk = blockIdx.x;
    const int b   = blk / P;
    const int p   = blk - b * P;
    const int tid = threadIdx.x;

    Wq += (size_t)p * sQKV;  Wk += (size_t)p * sQKV;  Wv += (size_t)p * sQKV;
    Wp += (size_t)p * sP;    bp += (size_t)p * sB;    qv += (size_t)p * sB;
    if (pWq) { pWq += (size_t)p * sPkQKV; pWk += (size_t)p * sPkQKV;
               pWv += (size_t)p * sPkQKV; pWp += (size_t)p * sPkP; }

    __shared__ __align__(16) unsigned short sX[LP * DPAD];   // x, later ctx
    __shared__ __align__(16) unsigned short sWt[304 * 40];   // slab staging
    constexpr int QKV_USH = 3 * LVAL * 304;
    constexpr int F_FLT   = LVAL * 208;
    constexpr int U_FLT   = ((QKV_USH + 1) / 2 > F_FLT) ? (QKV_USH + 1) / 2 : F_FLT;
    __shared__ __align__(16) float sU[U_FLT];                // q/k/v then ctx@Wp
    __shared__ float sA[LP];

    unsigned short* sQ = (unsigned short*)sU;
    unsigned short* sK = sQ + LVAL * 304;
    unsigned short* sV = sK + LVAL * 304;
    float*          sF = sU;

    // ---- Phase 1: load x -> LDS bf16, zero-padded to [LP][320] -------------
    for (int idx = tid; idx < LP * DPAD; idx += NTHREADS) {
        int row = idx / DPAD, col = idx - row * DPAD;
        float v = 0.f;
        if (row < LVAL && col < HD) {
            if constexpr (GATHER) {
                int tok = tokens[(size_t)blk * LVAL + row];
                v = emb[(size_t)tok * HD + col];
            } else {
                v = xin[((size_t)b * LVAL + row) * HD + col];
            }
        }
        sX[idx] = f2bfu(v);
    }
    __syncthreads();

    // ---- Phase 2: q/k/v projections (WMMA) --------------------------------
    gemm_bf16<LP, HD>(sX, sWt, Wq, pWq, sQ, nullptr, 304, LVAL, tid);
    gemm_bf16<LP, HD>(sX, sWt, Wk, pWk, sK, nullptr, 304, LVAL, tid);
    gemm_bf16<LP, HD>(sX, sWt, Wv, pWv, sV, nullptr, 304, LVAL, tid);

    // ---- Phase 3: per-head attention (VALU), ctx overwrites sX ------------
    const float rscale = 0.2236067977f;  // 1/sqrt(20)
    for (int item = tid; item < HEADS * LVAL; item += NTHREADS) {
        int h = item / LVAL;
        int m = item - h * LVAL;
        int cb = h * DKV;
        float qrow[DKV];
#pragma unroll
        for (int d = 0; d < DKV; ++d) qrow[d] = bf2f(sQ[m * 304 + cb + d]) * rscale;
        float s[LVAL];
        float mx = -1e30f;
        for (int m2 = 0; m2 < LVAL; ++m2) {
            float dot = 0.f;
#pragma unroll
            for (int d = 0; d < DKV; ++d) dot += qrow[d] * bf2f(sK[m2 * 304 + cb + d]);
            s[m2] = dot;
            mx = fmaxf(mx, dot);
        }
        float sum = 0.f;
        for (int m2 = 0; m2 < LVAL; ++m2) { float e = __expf(s[m2] - mx); s[m2] = e; sum += e; }
        float inv = 1.f / sum;
        float cacc[DKV];
#pragma unroll
        for (int d = 0; d < DKV; ++d) cacc[d] = 0.f;
        for (int m2 = 0; m2 < LVAL; ++m2) {
            float w = s[m2] * inv;
#pragma unroll
            for (int d = 0; d < DKV; ++d) cacc[d] += w * bf2f(sV[m2 * 304 + cb + d]);
        }
#pragma unroll
        for (int d = 0; d < DKV; ++d) sX[m * DPAD + cb + d] = f2bfu(cacc[d]);
    }
    __syncthreads();

    // ---- Phase 4: ctx @ Wp (WMMA) -> f32 in LDS (reuses q/k/v region) -----
    gemm_bf16<LP, AHID>(sX, sWt, Wp, pWp, nullptr, sF, 208, LVAL, tid);

    // ---- Phase 5: additive attention pooling ------------------------------
    if (tid < LVAL) {
        float dot = 0.f;
        for (int c = 0; c < AHID; ++c) {
            float t = tanhf(sF[tid * 208 + c] + bp[c]);
            dot += t * qv[c];
        }
        sA[tid] = dot;
    }
    __syncthreads();
    if (tid == 0) {
        float mx = -1e30f;
        for (int l = 0; l < LVAL; ++l) mx = fmaxf(mx, sA[l]);
        float sum = 0.f;
        for (int l = 0; l < LVAL; ++l) { float e = __expf(sA[l] - mx); sA[l] = e; sum += e; }
        float inv = 1.f / sum;
        for (int l = 0; l < LVAL; ++l) sA[l] *= inv;
    }
    __syncthreads();
    for (int c = tid; c < HD; c += NTHREADS) {
        float o = 0.f;
        for (int l = 0; l < LVAL; ++l) o += sA[l] * bf2f(sX[l * DPAD + c]);
        out[(size_t)blk * HD + c] = o;
    }
}

// ---------------------------------------------------------------------------
// scores = softmax(user_rep . cand_enc) over 5 candidates
// ---------------------------------------------------------------------------
__global__ __launch_bounds__(64) void score_kernel(
    const float* __restrict__ user,   // [B, 300]
    const float* __restrict__ cand,   // [B, 5, 300]
    float* __restrict__ outp)         // [B, 5]
{
    int b = blockIdx.x, t = threadIdx.x;
    __shared__ float s[5];
    if (t < 5) {
        float d = 0.f;
        for (int c = 0; c < HD; ++c)
            d += user[(size_t)b * HD + c] * cand[((size_t)b * 5 + t) * HD + c];
        s[t] = d;
    }
    __syncthreads();
    if (t < 5) {
        float v0 = s[0], v1 = s[1], v2 = s[2], v3 = s[3], v4 = s[4];
        float mx = fmaxf(fmaxf(fmaxf(v0, v1), fmaxf(v2, v3)), v4);
        float e0 = __expf(v0 - mx), e1 = __expf(v1 - mx), e2 = __expf(v2 - mx),
              e3 = __expf(v3 - mx), e4 = __expf(v4 - mx);
        float inv = 1.f / (e0 + e1 + e2 + e3 + e4);
        float mine = (t == 0) ? e0 : (t == 1) ? e1 : (t == 2) ? e2 : (t == 3) ? e3 : e4;
        outp[(size_t)b * 5 + t] = mine * inv;
    }
}

// ---------------------------------------------------------------------------
extern "C" void kernel_launch(void* const* d_in, const int* in_sizes, int n_in,
                              void* d_out, int out_size, void* d_ws, size_t ws_size,
                              hipStream_t stream) {
    (void)in_sizes; (void)n_in; (void)out_size;

    const int B = 128, NH = 50, NC = 5, L = 30;

    const int*   news_input = (const int*)d_in[0];    // [B,50,30]
    const int*   candidates = (const int*)d_in[1];    // [B,5,30]
    const float* emb        = (const float*)d_in[2];  // [60000,300]

    const float* his_Wq = (const float*)d_in[3];
    const float* his_Wk = (const float*)d_in[4];
    const float* his_Wv = (const float*)d_in[5];
    const float* his_Wp = (const float*)d_in[6];
    const float* his_bp = (const float*)d_in[7];
    const float* his_qv = (const float*)d_in[8];

    const float* cand_Wq = (const float*)d_in[9];
    const float* cand_Wk = (const float*)d_in[10];
    const float* cand_Wv = (const float*)d_in[11];
    const float* cand_Wp = (const float*)d_in[12];
    const float* cand_bp = (const float*)d_in[13];
    const float* cand_qv = (const float*)d_in[14];

    const float* usr_Wq = (const float*)d_in[15];
    const float* usr_Wk = (const float*)d_in[16];
    const float* usr_Wv = (const float*)d_in[17];
    const float* usr_Wp = (const float*)d_in[18];
    const float* usr_bp = (const float*)d_in[19];
    const float* usr_qv = (const float*)d_in[20];

    float* news_enc = (float*)d_ws;                   // [B,50,300]
    float* cand_enc = news_enc + (size_t)B * NH * HD; // [B,5,300]
    float* user_rep = cand_enc + (size_t)B * NC * HD; // [B,300]
    size_t actFloats = (size_t)B * NH * HD + (size_t)B * NC * HD + (size_t)B * HD;

    // Packed bf16 weight image sizes (ushorts)
    const size_t PK_QKV = 10 * 304 * 40;  // per matrix, N=300
    const size_t PK_P   = 10 * 208 * 40;  // per matrix, N=200
    const size_t pkTotal = (size_t)(NH + NC + 1) * (3 * PK_QKV + PK_P);
    const size_t needed  = actFloats * 4 + pkTotal * 2;

    unsigned short* pk = (unsigned short*)(user_rep + (size_t)B * HD);
    bool usePk = (ws_size >= needed);

    unsigned short *pHisQ = nullptr, *pHisK = nullptr, *pHisV = nullptr, *pHisP = nullptr;
    unsigned short *pCndQ = nullptr, *pCndK = nullptr, *pCndV = nullptr, *pCndP = nullptr;
    unsigned short *pUsrQ = nullptr, *pUsrK = nullptr, *pUsrV = nullptr, *pUsrP = nullptr;

    if (usePk) {
        unsigned short* q = pk;
        pHisQ = q; q += NH * PK_QKV;
        pHisK = q; q += NH * PK_QKV;
        pHisV = q; q += NH * PK_QKV;
        pHisP = q; q += NH * PK_P;
        pCndQ = q; q += NC * PK_QKV;
        pCndK = q; q += NC * PK_QKV;
        pCndV = q; q += NC * PK_QKV;
        pCndP = q; q += NC * PK_P;
        pUsrQ = q; q += PK_QKV;
        pUsrK = q; q += PK_QKV;
        pUsrV = q; q += PK_QKV;
        pUsrP = q;

        pack_kernel<HD><<<NH * 10, NTHREADS, 0, stream>>>(his_Wq, pHisQ, NH);
        pack_kernel<HD><<<NH * 10, NTHREADS, 0, stream>>>(his_Wk, pHisK, NH);
        pack_kernel<HD><<<NH * 10, NTHREADS, 0, stream>>>(his_Wv, pHisV, NH);
        pack_kernel<AHID><<<NH * 10, NTHREADS, 0, stream>>>(his_Wp, pHisP, NH);
        pack_kernel<HD><<<NC * 10, NTHREADS, 0, stream>>>(cand_Wq, pCndQ, NC);
        pack_kernel<HD><<<NC * 10, NTHREADS, 0, stream>>>(cand_Wk, pCndK, NC);
        pack_kernel<HD><<<NC * 10, NTHREADS, 0, stream>>>(cand_Wv, pCndV, NC);
        pack_kernel<AHID><<<NC * 10, NTHREADS, 0, stream>>>(cand_Wp, pCndP, NC);
        pack_kernel<HD><<<10, NTHREADS, 0, stream>>>(usr_Wq, pUsrQ, 1);
        pack_kernel<HD><<<10, NTHREADS, 0, stream>>>(usr_Wk, pUsrK, 1);
        pack_kernel<HD><<<10, NTHREADS, 0, stream>>>(usr_Wv, pUsrV, 1);
        pack_kernel<AHID><<<10, NTHREADS, 0, stream>>>(usr_Wp, pUsrP, 1);
    }

    const int sQKV = 300 * 300, sP = 300 * 200, sB = 200;

    encode_kernel<L, 32, true><<<B * NH, NTHREADS, 0, stream>>>(
        news_input, nullptr, emb,
        his_Wq, his_Wk, his_Wv, his_Wp, his_bp, his_qv,
        pHisQ, pHisK, pHisV, pHisP,
        NH, sQKV, sP, sB, (int)PK_QKV, (int)PK_P, news_enc);

    encode_kernel<L, 32, true><<<B * NC, NTHREADS, 0, stream>>>(
        candidates, nullptr, emb,
        cand_Wq, cand_Wk, cand_Wv, cand_Wp, cand_bp, cand_qv,
        pCndQ, pCndK, pCndV, pCndP,
        NC, sQKV, sP, sB, (int)PK_QKV, (int)PK_P, cand_enc);

    encode_kernel<NH, 64, false><<<B, NTHREADS, 0, stream>>>(
        nullptr, news_enc, emb,
        usr_Wq, usr_Wk, usr_Wv, usr_Wp, usr_bp, usr_qv,
        pUsrQ, pUsrK, pUsrV, pUsrP,
        1, 0, 0, 0, 0, 0, user_rep);

    score_kernel<<<B, 64, 0, stream>>>(user_rep, cand_enc, (float*)d_out);
}
